// GRUCell_47124381172356
// MI455X (gfx1250) — matrix operand order repaired and checked by reading, not compiled, not run on hardware
//
#include <hip/hip_runtime.h>
#include <hip/hip_bf16.h>
#include <math.h>

// ---------------------------------------------------------------------------
// GRU cell, fused: new_h = (1-z)*tanh(xn + r*(hn+b_hn)) + z*h
// B=16384, F=H=1024.  bf16 WMMA (v_wmma_f32_16x16x32_bf16), all 6 GEMM
// accumulations + gate math fused per 64x32 output tile.
// ---------------------------------------------------------------------------

typedef __bf16 bhalf;
typedef __attribute__((ext_vector_type(16))) __bf16 bf16x16;
typedef __attribute__((ext_vector_type(8)))  __bf16 bf16x8;
typedef __attribute__((ext_vector_type(8)))  float  f32x8;
typedef __attribute__((ext_vector_type(4)))  int    i32x4;

#define B_ROWS 16384
#define FEAT   1024
#define HID    1024
#define LDK    40      // padded LDS stride for a 32-wide K slab (bank spread)

// --- CDNA5 async global->LDS copy (guarded; falls back to sync copy) -------
#if defined(__gfx1250__) && __has_builtin(__builtin_amdgcn_global_load_async_to_lds_b128) && __has_builtin(__builtin_amdgcn_s_wait_asynccnt)
typedef __attribute__((address_space(1))) i32x4 gbl_i32x4;   // global int4
typedef __attribute__((address_space(3))) i32x4 lds_i32x4;   // LDS int4
#define CP16(dst, src) __builtin_amdgcn_global_load_async_to_lds_b128( \
    (gbl_i32x4*)(void*)(src), (lds_i32x4*)(dst), 0, 0)
#define CP_WAIT() __builtin_amdgcn_s_wait_asynccnt(0)
#else
#define CP16(dst, src) (*(bf16x8*)(dst) = *(const bf16x8*)(src))
#define CP_WAIT()
#endif

__device__ __forceinline__ bf16x16 frag16(const bhalf* p0, const bhalf* p1) {
    bf16x8 lo = *(const bf16x8*)p0;
    bf16x8 hi = *(const bf16x8*)p1;
    return __builtin_shufflevector(lo, hi, 0,1,2,3,4,5,6,7,8,9,10,11,12,13,14,15);
}

// --- pre-pass: f32 -> bf16 elementwise (x, h) ------------------------------
__global__ void cvt_f32_bf16_kernel(const float* __restrict__ src,
                                    bhalf* __restrict__ dst, int n) {
    int i = (blockIdx.x * blockDim.x + threadIdx.x) * 4;
    if (i + 3 < n) {
        const float4 v = *(const float4*)(src + i);
        dst[i+0] = (bhalf)v.x; dst[i+1] = (bhalf)v.y;
        dst[i+2] = (bhalf)v.z; dst[i+3] = (bhalf)v.w;
    }
}

// --- pre-pass: weights f32 [K][N] -> bf16 transposed [N][K] ----------------
__global__ void cvt_w_transpose_kernel(const float* w0, const float* w1,
                                       const float* w2, const float* w3,
                                       const float* w4, const float* w5,
                                       bhalf* __restrict__ dst) {
    const float* src;
    switch (blockIdx.z) {
        case 0: src = w0; break; case 1: src = w1; break;
        case 2: src = w2; break; case 3: src = w3; break;
        case 4: src = w4; break; default: src = w5; break;
    }
    int idx  = blockIdx.x * blockDim.x + threadIdx.x;   // 0..1048575
    int nrow = idx >> 10;
    int k    = idx & 1023;
    dst[(size_t)blockIdx.z * 1048576 + idx] = (bhalf)src[k * 1024 + nrow];
}

// --- fused GRU kernel ------------------------------------------------------
// grid (256, 32); block 256 (8 wave32s). WG tile 64 rows x 32 cols of new_h.
// Each wave: one 16x16 tile, 6 accumulators, K loop of 32 steps of K=32.
__global__ __launch_bounds__(256, 2)
void gru_wmma_kernel(const bhalf* __restrict__ xb, const bhalf* __restrict__ hb,
                     const bhalf* __restrict__ wb, const float* __restrict__ hsrc,
                     const float* __restrict__ b_ir, const float* __restrict__ b_iz,
                     const float* __restrict__ b_in, const float* __restrict__ b_hn,
                     float* __restrict__ out0, float* __restrict__ out1) {
    // LDS: 2 bufs x (X 64xLDK, H 64xLDK, W 6x32xLDK)  = 51,200 bytes
    __shared__ __align__(16) bhalf smem[2*64*LDK + 2*64*LDK + 2*6*32*LDK];
    bhalf* sX = smem;                 // [2][64][LDK]
    bhalf* sH = smem + 2*64*LDK;      // [2][64][LDK]
    bhalf* sW = smem + 4*64*LDK;      // [2][6][32][LDK]

    const int tid  = threadIdx.x;
    const int lane = tid & 31;
    const int wave = tid >> 5;
    const int wr   = wave & 3;        // 4 row-tiles of 16
    const int wc   = wave >> 2;       // 2 col-tiles of 16
    const int row0 = blockIdx.x * 64;
    const int col0 = blockIdx.y * 32;

    // staging indices: 256 threads, one b128 each for X and H, 3 for W
    const int st_row = tid >> 2;            // 0..63
    const int st_chk = (tid & 3) * 8;       // 0,8,16,24

#define LOAD_STAGE(b, kb) do {                                                  \
        CP16(&sX[(b)*2560 + st_row*LDK + st_chk],                               \
             &xb[(size_t)(row0 + st_row)*FEAT + (kb)*32 + st_chk]);             \
        CP16(&sH[(b)*2560 + st_row*LDK + st_chk],                               \
             &hb[(size_t)(row0 + st_row)*FEAT + (kb)*32 + st_chk]);             \
        _Pragma("unroll")                                                       \
        for (int i_ = 0; i_ < 3; ++i_) {                                        \
            int id_ = tid + i_*256;                                             \
            int j_  = id_ >> 7;            /* matrix 0..5 */                    \
            int rem_= id_ & 127;                                                \
            int nr_ = rem_ >> 2;           /* N row 0..31 */                    \
            int ck_ = (rem_ & 3) * 8;                                           \
            CP16(&sW[(b)*7680 + j_*32*LDK + nr_*LDK + ck_],                     \
                 &wb[(size_t)j_*1048576 + (size_t)(col0 + nr_)*1024             \
                     + (kb)*32 + ck_]);                                         \
        }                                                                       \
    } while (0)

    f32x8 axr = {0,0,0,0,0,0,0,0};
    f32x8 axz = axr, axn = axr, ahr = axr, ahz = axr, ahn = axr;

    // A fragment layout (16-bit A 16x32): lanes 0-15 -> M=lane, K {0..7,16..23};
    // lanes 16-31 -> M=lane-16, K {8..15,24..31}.
    const int am  = lane & 15;
    const int ak0 = (lane < 16) ? 0 : 8;
    // B fragment layout (16-bit B 32x16): lanes 0-15 -> N=lane, K {0..7,16..23};
    // lanes 16-31 -> N=lane-16, K {8..15,24..31}.  W stored [N][K].
    const int bn  = lane & 15;
    const int bk0 = (lane < 16) ? 0 : 8;

    LOAD_STAGE(0, 0);
    int buf = 0;
    for (int kb = 0; kb < 32; ++kb) {
        CP_WAIT();
        __syncthreads();
        if (kb + 1 < 32) LOAD_STAGE(buf ^ 1, kb + 1);

        const bhalf* pX = sX + buf*2560 + (wr*16 + am)*LDK;
        const bhalf* pH = sH + buf*2560 + (wr*16 + am)*LDK;
        bf16x16 ax = frag16(pX + ak0, pX + ak0 + 16);
        bf16x16 ah = frag16(pH + ak0, pH + ak0 + 16);

        const bhalf* pW = sW + buf*7680 + (wc*16 + bn)*LDK + bk0;
        bf16x16 bw;
        bw  = frag16(pW + 0*32*LDK, pW + 0*32*LDK + 16);
        axr = __builtin_amdgcn_wmma_f32_16x16x32_bf16(false, ax, false, bw, (short)0, axr, false, false);
        bw  = frag16(pW + 1*32*LDK, pW + 1*32*LDK + 16);
        axz = __builtin_amdgcn_wmma_f32_16x16x32_bf16(false, ax, false, bw, (short)0, axz, false, false);
        bw  = frag16(pW + 2*32*LDK, pW + 2*32*LDK + 16);
        axn = __builtin_amdgcn_wmma_f32_16x16x32_bf16(false, ax, false, bw, (short)0, axn, false, false);
        bw  = frag16(pW + 3*32*LDK, pW + 3*32*LDK + 16);
        ahr = __builtin_amdgcn_wmma_f32_16x16x32_bf16(false, ah, false, bw, (short)0, ahr, false, false);
        bw  = frag16(pW + 4*32*LDK, pW + 4*32*LDK + 16);
        ahz = __builtin_amdgcn_wmma_f32_16x16x32_bf16(false, ah, false, bw, (short)0, ahz, false, false);
        bw  = frag16(pW + 5*32*LDK, pW + 5*32*LDK + 16);
        ahn = __builtin_amdgcn_wmma_f32_16x16x32_bf16(false, ah, false, bw, (short)0, ahn, false, false);

        buf ^= 1;
    }
#undef LOAD_STAGE

    // Epilogue: C/D layout -> lane holds N = lane%16, VGPR r holds
    // M = r (lanes 0-15) / r+8 (lanes 16-31).
    const int m0 = wr*16 + ((lane < 16) ? 0 : 8);
    const int n  = col0 + wc*16 + (lane & 15);
    const float birv = b_ir[n], bizv = b_iz[n], binv = b_in[n], bhnv = b_hn[n];
#pragma unroll
    for (int r = 0; r < 8; ++r) {
        const int row  = row0 + m0 + r;
        const float hp = hsrc[(size_t)row * HID + n];
        const float rg = 1.0f / (1.0f + __expf(-(axr[r] + birv + ahr[r])));
        const float zg = 1.0f / (1.0f + __expf(-(axz[r] + bizv + ahz[r])));
        const float ng = tanhf(axn[r] + binv + rg * (ahn[r] + bhnv));
        const float nh = (1.0f - zg) * ng + zg * hp;
        out0[(size_t)row * HID + n] = nh;
        out1[(size_t)row * HID + n] = nh;
    }
}

// ---------------------------------------------------------------------------
extern "C" void kernel_launch(void* const* d_in, const int* in_sizes, int n_in,
                              void* d_out, int out_size, void* d_ws, size_t ws_size,
                              hipStream_t stream) {
    const float* h   = (const float*)d_in[0];
    const float* x   = (const float*)d_in[1];
    const float* Wir = (const float*)d_in[2];
    const float* Wiz = (const float*)d_in[3];
    const float* Win = (const float*)d_in[4];
    const float* bir = (const float*)d_in[5];
    const float* biz = (const float*)d_in[6];
    const float* bin = (const float*)d_in[7];
    const float* Whr = (const float*)d_in[8];
    const float* Whz = (const float*)d_in[9];
    const float* Whn = (const float*)d_in[10];
    const float* bhn = (const float*)d_in[11];

    // workspace layout (bf16 elems): xb[16M] | hb[16M] | wb[6M transposed]
    bhalf* ws = (bhalf*)d_ws;
    bhalf* xb = ws;
    bhalf* hb = ws + (size_t)16777216;
    bhalf* wb = ws + (size_t)33554432;

    float* out0 = (float*)d_out;                       // new_h (tuple elem 0)
    float* out1 = out0 + (size_t)B_ROWS * HID;         // new_h (tuple elem 1)

    cvt_f32_bf16_kernel<<<16384, 256, 0, stream>>>(x, xb, B_ROWS * FEAT);
    cvt_f32_bf16_kernel<<<16384, 256, 0, stream>>>(h, hb, B_ROWS * HID);
    cvt_w_transpose_kernel<<<dim3(4096, 1, 6), 256, 0, stream>>>(
        Wir, Wiz, Win, Whr, Whz, Whn, wb);
    gru_wmma_kernel<<<dim3(B_ROWS / 64, HID / 32), 256, 0, stream>>>(
        xb, hb, wb, h, bir, biz, bin, bhn, out0, out1);
}
